// LogFlashAttentionNd_10127532884569
// MI455X (gfx1250) — matrix operand order repaired and checked by reading, not compiled, not run on hardware
//
#include <hip/hip_runtime.h>

typedef _Float16 f16;
typedef __attribute__((ext_vector_type(16))) _Float16 v16h;
typedef __attribute__((ext_vector_type(8)))  _Float16 v8h;
typedef __attribute__((ext_vector_type(8)))  float    v8f;

#define BB  4
#define CC  512
#define WW  4096
#define HH  8
#define DDIM 64
#define LP5 15625
#define C2  (2*CC)
#define BTS 40   // Bt row stride in f16 (80B: 16B-aligned rows, conflict-free frag reads)

// ---------------------------------------------------------------------------
// Elementwise helpers
// ---------------------------------------------------------------------------
__global__ void k_cvt_f32_f16(const float* __restrict__ in, f16* __restrict__ out, long n) {
  long i = (long)blockIdx.x * blockDim.x + threadIdx.x;
  if (i < n) out[i] = (f16)in[i];
}

__global__ void k_copy_f32(const float* __restrict__ in, float* __restrict__ out, long n) {
  long i = (long)blockIdx.x * blockDim.x + threadIdx.x;
  if (i < n) out[i] = in[i];
}

__global__ void k_zero_f32(float* __restrict__ out, long n) {
  long i = (long)blockIdx.x * blockDim.x + threadIdx.x;
  if (i < n) out[i] = 0.0f;
}

// y[b,c,w] += v_src[b,c,w]   (v_src may have a wider row stride: Lp buffer)
__global__ void k_add_att(float* __restrict__ y, const float* __restrict__ v,
                          long vs, long vbs) {
  long i = (long)blockIdx.x * blockDim.x + threadIdx.x;
  if (i >= (long)BB * CC * WW) return;
  int wp = (int)(i % WW);
  int c  = (int)((i / WW) % CC);
  int b  = (int)(i / ((long)WW * CC));
  y[i] += v[(long)b * vbs + (long)c * vs + wp];
}

// y += affine(dwconv3(v, wpe), spe, bpe)
__global__ void k_pe_acc(const float* __restrict__ v, const float* __restrict__ wpe,
                         const float* __restrict__ spe, const float* __restrict__ bpe,
                         float* __restrict__ y, long vs, long vbs) {
  long i = (long)blockIdx.x * blockDim.x + threadIdx.x;
  if (i >= (long)BB * CC * WW) return;
  int wp = (int)(i % WW);
  int c  = (int)((i / WW) % CC);
  int b  = (int)(i / ((long)WW * CC));
  const float* vr = v + (long)b * vbs + (long)c * vs;
  float lft = (wp > 0)      ? vr[wp - 1] : 0.0f;
  float mid = vr[wp];
  float rgt = (wp < WW - 1) ? vr[wp + 1] : 0.0f;
  float cv  = wpe[c * 3 + 0] * lft + wpe[c * 3 + 1] * mid + wpe[c * 3 + 2] * rgt;
  y[i] += spe[c] * cv + bpe[c];
}

// ---------------------------------------------------------------------------
// WMMA GEMM:  out[b,oc,n] = affine( sum_c Wt[oc,c] * X[b,c,n] )
// 128 threads = 4 waves; block tile 128(M) x 64(N); K stepped by 32.
// Each wave owns a 32x64 strip: 2 A-fragments, 8 accumulators; every
// B-fragment read feeds two WMMAs. Output flavor is a template parameter so
// the epilogue has no per-element branches.
// ---------------------------------------------------------------------------
template <bool HASF>
__global__ __launch_bounds__(128)
void k_gemm_f16(const f16* __restrict__ Wt,
                const f16* __restrict__ X,
                const float* __restrict__ scale, const float* __restrict__ shift,
                float* __restrict__ outF, f16* __restrict__ outH,
                long ostride, long obstride) {
  __shared__ f16 As[128 * 32];   // A tile: [m][k], rows 64B
  __shared__ f16 Bt[64 * BTS];   // B tile transposed: [n][k], rows 80B
  const int tid = threadIdx.x;
  const int wv  = tid >> 5;
  const int ln  = tid & 31;
  const int hf  = ln >> 4;       // 0/1 : which 16-lane half
  const int nl  = ln & 15;
  const int b   = blockIdx.z;
  const long tileN = (long)blockIdx.x * 64;
  const int  tileM = blockIdx.y * 128;
  const f16* Xb = X + (long)b * CC * WW;

  // per-thread staging chunk coordinates (constant across K steps)
  int am[4], ak[4];
#pragma unroll
  for (int i = 0; i < 4; ++i) {                   // A: 512 chunks of 8
    int c = tid + i * 128;
    am[i] = c >> 2; ak[i] = (c & 3) * 8;
  }
  const int bk0 = tid >> 3,        bn0 = (tid & 7) * 8;          // B: 256 chunks
  const int bk1 = (tid + 128) >> 3, bn1 = ((tid + 128) & 7) * 8;

  v8f acc0[4] = {v8f{}, v8f{}, v8f{}, v8f{}};
  v8f acc1[4] = {v8f{}, v8f{}, v8f{}, v8f{}};

  for (int kk = 0; kk < CC; kk += 32) {
    __syncthreads();
    // ---- stage A (vector both sides) ----
    v8h a[4];
#pragma unroll
    for (int i = 0; i < 4; ++i)
      a[i] = *(const v8h*)&Wt[(long)(tileM + am[i]) * CC + kk + ak[i]];
    // ---- stage B: vector global loads, register transpose, LDS scatter ----
    v8h b0 = *(const v8h*)&Xb[(long)(kk + bk0) * WW + tileN + bn0];
    v8h b1 = *(const v8h*)&Xb[(long)(kk + bk1) * WW + tileN + bn1];
    if (kk + 32 < CC)                             // -> global_prefetch_b8
      __builtin_prefetch(&Xb[(long)(kk + 32) * WW + tileN + (tid & 63)], 0, 0);
#pragma unroll
    for (int i = 0; i < 4; ++i)
      *(v8h*)&As[am[i] * 32 + ak[i]] = a[i];
#pragma unroll
    for (int e = 0; e < 8; ++e) {
      Bt[(bn0 + e) * BTS + bk0] = b0[e];
      Bt[(bn1 + e) * BTS + bk1] = b1[e];
    }
    __syncthreads();

    // two A fragments: rows wv*32+nl and wv*32+16+nl (ISA 16-bit A layout)
    v16h af0, af1;
    {
      int m0 = wv * 32 + nl;
      int m1 = m0 + 16;
#pragma unroll
      for (int j = 0; j < 8; ++j) {
        af0[j]     = As[m0 * 32 + hf * 8 + j];
        af0[8 + j] = As[m0 * 32 + 16 + hf * 8 + j];
        af1[j]     = As[m1 * 32 + hf * 8 + j];
        af1[8 + j] = As[m1 * 32 + 16 + hf * 8 + j];
      }
    }
#pragma unroll
    for (int t = 0; t < 4; ++t) {                 // each B-frag feeds 2 WMMAs
      v16h bf;
      int n = t * 16 + nl;
#pragma unroll
      for (int e = 0; e < 16; ++e) bf[e] = Bt[n * BTS + hf * 8 + e + (e < 8 ? 0 : 8)];
      acc0[t] = __builtin_amdgcn_wmma_f32_16x16x32_f16(false, af0, false, bf,
                                                       (short)0, acc0[t], false, false);
      acc1[t] = __builtin_amdgcn_wmma_f32_16x16x32_f16(false, af1, false, bf,
                                                       (short)0, acc1[t], false, false);
    }
  }

#if __has_builtin(__builtin_amdgcn_s_wait_tensorcnt)
  __builtin_amdgcn_s_wait_tensorcnt(0);           // gfx1250 split-counter path
#endif

#pragma unroll
  for (int r = 0; r < 2; ++r) {
    const v8f* acc = r ? acc1 : acc0;
#pragma unroll
    for (int j = 0; j < 8; ++j) {
      int oc = tileM + wv * 32 + r * 16 + hf * 8 + j;  // D layout: M=(half?8:0)+j
      float sc = scale[oc], sh = shift[oc];
#pragma unroll
      for (int t = 0; t < 4; ++t) {
        float vo = acc[t][j] * sc + sh;
        long off = (long)b * obstride + (long)oc * ostride + tileN + t * 16 + nl;
        if (HASF) outF[off] = vo;
        else      outH[off] = (f16)vo;
      }
    }
  }
}

// ---------------------------------------------------------------------------
// Base-16 hierarchical attention round, in place on v (f32, [B,C,W]).
// One block per (group, batch); 8 waves = 8 heads. Group members are
// p = p0 + j*s, j=0..15.
// ---------------------------------------------------------------------------
__global__ __launch_bounds__(256)
void k_attn16(const f16* __restrict__ qk, float* __restrict__ v, int s) {
  __shared__ f16 Qs[HH][16][DDIM];
  __shared__ f16 Ks[HH][16][DDIM];
  __shared__ f16 Vs[HH][16][DDIM];
  __shared__ f16 Ps[HH][16][16];
  const int tid = threadIdx.x;
  const int h   = tid >> 5;
  const int ln  = tid & 31;
  const int hf  = ln >> 4;
  const int nl  = ln & 15;
  const int g   = blockIdx.x;
  const int b   = blockIdx.y;
  const long p0 = (long)(g / s) * (s * 16) + (g % s);
  const f16* qb = qk + (long)b * C2 * WW;
  float*     vb = v  + (long)b * CC * WW;

  for (int i = 0; i < 32; ++i) {                 // stage Q,K,V for this head
    int lin = i * 32 + ln;                       // 0..1023
    int m = lin >> 6, dd = lin & 63;
    long pos = p0 + (long)m * s;
    int ch = h * DDIM + dd;
    Qs[h][m][dd] = qb[(long)ch * WW + pos];
    Ks[h][m][dd] = qb[(long)(CC + ch) * WW + pos];
    Vs[h][m][dd] = (f16)vb[(long)ch * WW + pos];
  }
  __syncthreads();

  v8f sacc = {};                                  // S = Q K^T  (K depth 64 = 2x32)
#pragma unroll
  for (int kc = 0; kc < 2; ++kc) {
    v16h af, bf;
#pragma unroll
    for (int j = 0; j < 8; ++j) {
      af[j]     = Qs[h][nl][kc * 32 + hf * 8 + j];
      af[8 + j] = Qs[h][nl][kc * 32 + 16 + hf * 8 + j];
    }
#pragma unroll
    for (int e = 0; e < 16; ++e) bf[e] = Ks[h][nl][kc * 32 + hf * 8 + e + (e < 8 ? 0 : 8)];
    sacc = __builtin_amdgcn_wmma_f32_16x16x32_f16(false, af, false, bf,
                                                  (short)0, sacc, false, false);
  }

  // softmax over keys: lane holds S[hf*8+j][nl]; reduce across 16-lane half
  float p[8];
#pragma unroll
  for (int j = 0; j < 8; ++j) p[j] = sacc[j] * 0.125f;   // 1/sqrt(64)
#pragma unroll
  for (int j = 0; j < 8; ++j) {
    float mx = p[j];
    for (int d = 1; d < 16; d <<= 1) mx = fmaxf(mx, __shfl_xor(mx, d, 32));
    float e = __expf(p[j] - mx);
    float sm = e;
    for (int d = 1; d < 16; d <<= 1) sm += __shfl_xor(sm, d, 32);
    p[j] = e / sm;
  }
#pragma unroll
  for (int j = 0; j < 8; ++j) Ps[h][hf * 8 + j][nl] = (f16)p[j];
  __syncthreads();

  v16h pf;                                        // P as A-fragment, K padded to 32
#pragma unroll
  for (int e = 0; e < 16; ++e) {
    int K = (e < 8 ? e : e + 8) + (hf ? 8 : 0);
    pf[e] = (K < 16) ? Ps[h][nl][K] : (f16)0.0f;
  }
#pragma unroll
  for (int t = 0; t < 4; ++t) {                   // O = P @ V, 4 N-tiles of 16
    v16h bf;
#pragma unroll
    for (int e = 0; e < 16; ++e) {
      int K = hf * 8 + e + (e < 8 ? 0 : 8);
      bf[e] = (K < 16) ? Vs[h][K][t * 16 + nl] : (f16)0.0f;
    }
    v8f oacc = {};
    oacc = __builtin_amdgcn_wmma_f32_16x16x32_f16(false, pf, false, bf,
                                                  (short)0, oacc, false, false);
#pragma unroll
    for (int j = 0; j < 8; ++j) {
      long pos = p0 + (long)(hf * 8 + j) * s;
      int ch = h * DDIM + t * 16 + nl;
      vb[(long)ch * WW + pos] = oacc[j];
    }
  }
}

// ---------------------------------------------------------------------------
// Base-5 attention round on padded Lp buffer (VALU path; 5x5 tiles are too
// small for 16x16 WMMA). One 64-thread block per (group, head, batch);
// thread t owns feature dim t.
// ---------------------------------------------------------------------------
__global__ __launch_bounds__(64)
void k_attn5(const float* __restrict__ qk, float* __restrict__ v, int s) {
  __shared__ float red[2][25];
  const int t   = threadIdx.x;
  const int wid = t >> 5, ln = t & 31;
  const int g   = blockIdx.x;
  const int h   = blockIdx.y;
  const int b   = blockIdx.z;
  const long p0 = (long)(g / s) * (s * 5) + (g % s);
  const float* qb = qk + (long)b * C2 * WW + (long)(h * DDIM + t) * WW;
  const float* kb = qb + (long)CC * WW;
  float*       vb = v  + (long)b * CC * LP5 + (long)(h * DDIM + t) * LP5;

  float q[5], k[5], vv[5];
#pragma unroll
  for (int j = 0; j < 5; ++j) {
    long pos = p0 + (long)j * s;
    bool in = pos < WW;                 // padded positions: q=k=0 (matches ref)
    q[j]  = in ? qb[pos] : 0.0f;
    k[j]  = in ? kb[pos] : 0.0f;
    vv[j] = vb[pos];
  }
  float sc[25];
#pragma unroll
  for (int i = 0; i < 5; ++i)
#pragma unroll
    for (int j = 0; j < 5; ++j) {
      float x = q[i] * k[j];
      for (int d = 1; d < 32; d <<= 1) x += __shfl_xor(x, d, 32);
      sc[i * 5 + j] = x;
    }
  if (ln == 0) {
#pragma unroll
    for (int i = 0; i < 25; ++i) red[wid][i] = sc[i];
  }
  __syncthreads();
#pragma unroll
  for (int i = 0; i < 25; ++i) sc[i] = (red[0][i] + red[1][i]) * 0.125f;

  float p[25];
#pragma unroll
  for (int i = 0; i < 5; ++i) {
    float mx = sc[i * 5];
    for (int j = 1; j < 5; ++j) mx = fmaxf(mx, sc[i * 5 + j]);
    float sm = 0.0f;
    for (int j = 0; j < 5; ++j) { p[i * 5 + j] = __expf(sc[i * 5 + j] - mx); sm += p[i * 5 + j]; }
    float inv = 1.0f / sm;
    for (int j = 0; j < 5; ++j) p[i * 5 + j] *= inv;
  }
#pragma unroll
  for (int i = 0; i < 5; ++i) {
    float o = 0.0f;
#pragma unroll
    for (int j = 0; j < 5; ++j) o += p[i * 5 + j] * vv[j];
    vb[p0 + (long)i * s] = o;
  }
}

// ---------------------------------------------------------------------------
// Host launcher
// ---------------------------------------------------------------------------
static inline size_t alignUp(size_t x) { return (x + 255) & ~(size_t)255; }

extern "C" void kernel_launch(void* const* d_in, const int* in_sizes, int n_in,
                              void* d_out, int out_size, void* d_ws, size_t ws_size,
                              hipStream_t stream) {
  (void)in_sizes; (void)n_in; (void)out_size; (void)ws_size;
  const float* x     = (const float*)d_in[0];
  const float* wqk   = (const float*)d_in[1];
  const float* sqk   = (const float*)d_in[2];
  const float* bqk   = (const float*)d_in[3];
  const float* wv    = (const float*)d_in[4];
  const float* sv    = (const float*)d_in[5];
  const float* bv    = (const float*)d_in[6];
  const float* wpe   = (const float*)d_in[7];
  const float* spe   = (const float*)d_in[8];
  const float* bpe   = (const float*)d_in[9];
  const float* wproj = (const float*)d_in[10];
  const float* sproj = (const float*)d_in[11];
  const float* bproj = (const float*)d_in[12];
  float* out = (float*)d_out;

  const long NBCW = (long)BB * CC * WW;           // 8,388,608
  const long NQK  = (long)BB * C2 * WW;           // 16,777,216
  const long NV5  = (long)BB * CC * LP5;          // 32,000,000

  char* ws = (char*)d_ws; size_t off = 0;
  auto alloc = [&](size_t bytes) { void* p = ws + off; off = alignUp(off + bytes); return p; };
  f16*   xb     = (f16*)  alloc(NBCW * sizeof(f16));
  f16*   wqkb   = (f16*)  alloc((size_t)2 * C2 * CC * sizeof(f16));
  f16*   wvb    = (f16*)  alloc((size_t)2 * CC * CC * sizeof(f16));
  f16*   wprojb = (f16*)  alloc((size_t)CC * CC * sizeof(f16));
  f16*   qk16   = (f16*)  alloc(NQK  * sizeof(f16));
  float* v0     = (float*)alloc(NBCW * sizeof(float));
  float* qk5    = (float*)alloc(NQK  * sizeof(float));
  float* v1     = (float*)alloc(NV5  * sizeof(float));
  float* yacc   = (float*)alloc(NBCW * sizeof(float));
  f16*   yb     = (f16*)  alloc(NBCW * sizeof(f16));

  auto gblk = [](long n) { return (unsigned)((n + 255) / 256); };

  // conversions + init
  k_cvt_f32_f16<<<gblk(NBCW), 256, 0, stream>>>(x, xb, NBCW);
  k_cvt_f32_f16<<<gblk((long)2 * C2 * CC), 256, 0, stream>>>(wqk, wqkb, (long)2 * C2 * CC);
  k_cvt_f32_f16<<<gblk((long)2 * CC * CC), 256, 0, stream>>>(wv, wvb, (long)2 * CC * CC);
  k_cvt_f32_f16<<<gblk((long)CC * CC), 256, 0, stream>>>(wproj, wprojb, (long)CC * CC);
  k_copy_f32<<<gblk(NBCW), 256, 0, stream>>>(x, yacc, NBCW);
  k_zero_f32<<<gblk(NV5), 256, 0, stream>>>(v1, NV5);

  dim3 gemmB(128);
  // ---- base 16 block ----
  k_gemm_f16<false><<<dim3(WW / 64, C2 / 128, BB), gemmB, 0, stream>>>(
      wqkb, xb, sqk, bqk, nullptr, qk16, (long)WW, (long)C2 * WW);
  k_gemm_f16<true><<<dim3(WW / 64, CC / 128, BB), gemmB, 0, stream>>>(
      wvb, xb, sv, bv, v0, nullptr, (long)WW, (long)CC * WW);
  k_pe_acc<<<gblk(NBCW), 256, 0, stream>>>(v0, wpe, spe, bpe, yacc, (long)WW, (long)CC * WW);
  k_attn16<<<dim3(WW / 16, BB), 256, 0, stream>>>(qk16, v0, 256);
  k_attn16<<<dim3(WW / 16, BB), 256, 0, stream>>>(qk16, v0, 16);
  k_attn16<<<dim3(WW / 16, BB), 256, 0, stream>>>(qk16, v0, 1);
  k_add_att<<<gblk(NBCW), 256, 0, stream>>>(yacc, v0, (long)WW, (long)CC * WW);

  // ---- base 5 block ----
  k_gemm_f16<true><<<dim3(WW / 64, C2 / 128, BB), gemmB, 0, stream>>>(
      wqkb + (size_t)C2 * CC, xb, sqk + C2, bqk + C2, qk5, nullptr,
      (long)WW, (long)C2 * WW);
  k_gemm_f16<true><<<dim3(WW / 64, CC / 128, BB), gemmB, 0, stream>>>(
      wvb + (size_t)CC * CC, xb, sv + CC, bv + CC, v1, nullptr,
      (long)LP5, (long)CC * LP5);
  k_pe_acc<<<gblk(NBCW), 256, 0, stream>>>(v1, wpe + (size_t)CC * 3, spe + CC, bpe + CC,
                                           yacc, (long)LP5, (long)CC * LP5);
  const int strides5[6] = {3125, 625, 125, 25, 5, 1};
  for (int r = 0; r < 6; ++r)
    k_attn5<<<dim3(LP5 / 5, HH, BB), 64, 0, stream>>>(qk5, v1, strides5[r]);
  k_add_att<<<gblk(NBCW), 256, 0, stream>>>(yacc, v1, (long)LP5, (long)CC * LP5);

  // ---- final projection ----
  k_cvt_f32_f16<<<gblk(NBCW), 256, 0, stream>>>(yacc, yb, NBCW);
  k_gemm_f16<true><<<dim3(WW / 64, CC / 128, BB), gemmB, 0, stream>>>(
      wprojb, yb, sproj, bproj, out, nullptr, (long)WW, (long)CC * WW);
}
